// MViTBlock_58317065945631
// MI455X (gfx1250) — matrix operand order repaired and checked by reading, not compiled, not run on hardware
//
#include <hip/hip_runtime.h>
#include <hip/hip_bf16.h>
#include <math.h>

typedef __bf16 bf;
typedef __attribute__((ext_vector_type(16))) __bf16 v16bf;
typedef __attribute__((ext_vector_type(8)))  float   v8f;
typedef __attribute__((ext_vector_type(4)))  unsigned int u32x4;
typedef __attribute__((ext_vector_type(4)))  int         i32x4;

union FragB { v16bf v; u32x4 u[2]; bf e[16]; };

__device__ __forceinline__ v8f wmma_bf16(v16bf a, v16bf b, v8f c) {
  // D = A(16x32 bf16) * B(32x16 bf16) + C(16x16 f32)
  return __builtin_amdgcn_wmma_f32_16x16x32_bf16(false, a, false, b, (short)0, c, false, false);
}

__device__ __forceinline__ float gelu_exact(float v) {
  return 0.5f * v * (1.0f + erff(v * 0.70710678118654752f));
}

// ---- async Global->LDS (gfx1250) with graceful fallback --------------------
#if defined(__HIP_DEVICE_COMPILE__) && __has_builtin(__builtin_amdgcn_global_load_async_to_lds_b128)
#define HAVE_ASYNC_LDS 1
#endif

__device__ __forceinline__ void async_copy16(const bf* g, bf* l) {
#if defined(HAVE_ASYNC_LDS)
  // builtin signature: (V4i addrspace(1)*, V4i addrspace(3)*, imm offset, imm cpol)
  __builtin_amdgcn_global_load_async_to_lds_b128(
      (__attribute__((address_space(1))) i32x4*)g,
      (__attribute__((address_space(3))) i32x4*)l, 0, 0);
#else
  *(u32x4*)l = *(const u32x4*)g;
#endif
}

__device__ __forceinline__ void wait_async0() {
#if defined(HAVE_ASYNC_LDS)
#if __has_builtin(__builtin_amdgcn_s_wait_asynccnt)
  __builtin_amdgcn_s_wait_asynccnt(0);
#else
  asm volatile("s_wait_asynccnt 0x0" ::: "memory");
#endif
#endif
}

// ---------- weight convert + transpose: Wt[n][k] = (bf16)W[k][n] ----------
__global__ void k_cvt_transpose(const float* __restrict__ W, bf* __restrict__ Wt,
                                int K, int Ncol) {
  int idx = blockIdx.x * blockDim.x + threadIdx.x;
  int total = K * Ncol;
  if (idx >= total) return;
  int k = idx / Ncol, n = idx % Ncol;
  Wt[(size_t)n * K + k] = (bf)W[idx];
}

// ---------- layernorm row -> bf16 ----------
__global__ __launch_bounds__(128) void k_layernorm_bf16(
    const float* __restrict__ x, const float* __restrict__ w,
    const float* __restrict__ bia, bf* __restrict__ y, int C) {
  int row = blockIdx.x;
  const float* xr = x + (size_t)row * C;
  float s = 0.f, s2 = 0.f;
  for (int c = threadIdx.x; c < C; c += blockDim.x) {
    float v = xr[c]; s += v; s2 += v * v;
  }
  __shared__ float rs[4], rs2[4];
  for (int o = 16; o; o >>= 1) { s += __shfl_down(s, o, 32); s2 += __shfl_down(s2, o, 32); }
  int wid = threadIdx.x >> 5, lane = threadIdx.x & 31;
  if (lane == 0) { rs[wid] = s; rs2[wid] = s2; }
  __syncthreads();
  s = rs[0] + rs[1] + rs[2] + rs[3];
  s2 = rs2[0] + rs2[1] + rs2[2] + rs2[3];
  float mu = s / C;
  float var = s2 / C - mu * mu;
  float inv = rsqrtf(var + 1e-6f);
  bf* yr = y + (size_t)row * C;
  for (int c = threadIdx.x; c < C; c += blockDim.x)
    yr[c] = (bf)((xr[c] - mu) * inv * w[c] + bia[c]);
}

// ---------- 2x2 avg pool of raw x -> bf16 kv tokens ----------
__global__ void k_pool_bf16(const float* __restrict__ x, bf* __restrict__ kv,
                            int Bn, int Hh, int Ww, int C) {
  int idx = blockIdx.x * blockDim.x + threadIdx.x;
  int Hk = Hh >> 1, Wk = Ww >> 1;
  int total = Bn * Hk * Wk * C;
  if (idx >= total) return;
  int c = idx % C; int t = idx / C;
  int wk = t % Wk; t /= Wk;
  int hk = t % Hk; int b = t / Hk;
  const float* xb = x + ((size_t)b * Hh * Ww) * C;
  int r0 = (2 * hk) * Ww + 2 * wk;
  int r1 = r0 + Ww;
  float v = 0.25f * (xb[(size_t)r0 * C + c] + xb[(size_t)(r0 + 1) * C + c] +
                     xb[(size_t)r1 * C + c] + xb[(size_t)(r1 + 1) * C + c]);
  kv[idx] = (bf)v;
}

// ---------- block-tiled bf16 WMMA GEMM ----------
// Block = 256 threads (8 waves) computes 128(M) x 64(N).
// Wave w: rows m0+16w..+15, 4 accumulators over the 64-col panel.
// B panel (64 cols x 32 k, bf16) staged in LDS via async Global->LDS,
// double buffered; A fragments streamed straight from global (b128/lane).
// A: [M][K] bf16 row-major.  Wt: [Ncol][K] bf16 (W transposed).
// layout 0: plain [M][Ncol]; 1: [b][head][r][d]; 2: [b][head][d][r]
__global__ __launch_bounds__(256) void k_gemm_wmma(
    const bf* __restrict__ A, const bf* __restrict__ Wt,
    const float* __restrict__ bias, const float* __restrict__ resid,
    float* __restrict__ outf, bf* __restrict__ outb,
    int M, int Ncol, int K, int layout, int rows_per_batch, int head_dim, int act) {
  __shared__ bf panel[2][64 * 32];

  int gn = Ncol >> 6;
  int m0 = (blockIdx.x / gn) << 7;
  int n0 = (blockIdx.x % gn) << 6;
  int wave = threadIdx.x >> 5;
  int lane = threadIdx.x & 31;
  int lo = lane & 15, lhi = lane >> 4;
  int kb = lhi << 3;

  // staging assignment: thread t copies 16B: col = t>>2, chunk = t&3
  int sc_ = threadIdx.x >> 2, sch = threadIdx.x & 3;
  const bf* srow = Wt + (size_t)(n0 + sc_) * K + sch * 8;
  bf* sdst = &panel[0][sc_ * 32 + sch * 8];

  const bf* arow = A + (size_t)(m0 + (wave << 4) + lo) * K + (lhi << 4);

  v8f acc0 = {}, acc1 = {}, acc2 = {}, acc3 = {};
  int KC = K >> 5;

  async_copy16(srow, sdst);           // panel 0
  wait_async0();
  __syncthreads();

  for (int kc = 0; kc < KC; ++kc) {
    int buf = kc & 1;
    if (kc + 1 < KC)                  // prefetch next panel into the free buffer
      async_copy16(srow + ((kc + 1) << 5), &panel[buf ^ 1][sc_ * 32 + sch * 8]);

    FragB a;
    a.u[0] = *(const u32x4*)(arow + (kc << 5));
    a.u[1] = *(const u32x4*)(arow + (kc << 5) + 8);

    const bf* pb_ = &panel[buf][lo * 32 + (lhi << 4)];
    FragB b0, b1, b2, b3;
    b0.u[0] = *(const u32x4*)(pb_);           b0.u[1] = *(const u32x4*)(pb_ + 8);
    b1.u[0] = *(const u32x4*)(pb_ + 512);     b1.u[1] = *(const u32x4*)(pb_ + 520);
    b2.u[0] = *(const u32x4*)(pb_ + 1024);    b2.u[1] = *(const u32x4*)(pb_ + 1032);
    b3.u[0] = *(const u32x4*)(pb_ + 1536);    b3.u[1] = *(const u32x4*)(pb_ + 1544);
    acc0 = wmma_bf16(a.v, b0.v, acc0);
    acc1 = wmma_bf16(a.v, b1.v, acc1);
    acc2 = wmma_bf16(a.v, b2.v, acc2);
    acc3 = wmma_bf16(a.v, b3.v, acc3);

    if (kc + 1 < KC) {
      wait_async0();                  // our slice of next panel landed
      __syncthreads();                // everyone's slice landed & buf free
    }
  }

  int mw = m0 + (wave << 4);
  v8f accs[4] = {acc0, acc1, acc2, acc3};
#pragma unroll
  for (int nt = 0; nt < 4; ++nt) {
    int ncol = n0 + (nt << 4) + lo;
    float bia = bias ? bias[ncol] : 0.f;
#pragma unroll
    for (int i = 0; i < 8; ++i) {
      int row = mw + i + (lhi << 3);
      float v = accs[nt][i] + bia;
      if (act) v = gelu_exact(v);
      if (resid) v += resid[(size_t)row * Ncol + ncol];
      size_t oidx;
      if (layout == 0) {
        oidx = (size_t)row * Ncol + ncol;
      } else {
        int bidx = row / rows_per_batch, r = row % rows_per_batch;
        int head = ncol / head_dim, dd = ncol % head_dim;
        int nheads = Ncol / head_dim;
        if (layout == 1)
          oidx = (((size_t)bidx * nheads + head) * rows_per_batch + r) * head_dim + dd;
        else
          oidx = (((size_t)bidx * nheads + head) * head_dim + dd) * rows_per_batch + r;
      }
      if (outf) outf[oidx] = v;
      if (outb) outb[oidx] = (bf)v;
    }
  }
}

// ---------- attention: S = softmax(Q K^T * scale); O = S V ----------
// Q:[b][h][NQ][48], K:[b][h][NK][48], Vt:[b][h][48][NK]; O:[b][NQ][C]
__global__ __launch_bounds__(64) void k_attn_wmma(
    const bf* __restrict__ Q, const bf* __restrict__ Kc, const bf* __restrict__ Vt,
    bf* __restrict__ O, int NQ, int NKv, int HD, int NH, int Cc, float scale) {
  extern __shared__ float smem[];
  int wave = threadIdx.x >> 5, lane = threadIdx.x & 31;
  int lo = lane & 15, lhi = lane >> 4, kb = lhi << 3;
  int q0 = (blockIdx.x * 2 + wave) << 4;
  int head = blockIdx.y, b = blockIdx.z;
  const bf* Qh = Q  + (((size_t)b * NH + head) * NQ) * HD;
  const bf* Kh = Kc + (((size_t)b * NH + head) * NKv) * HD;
  const bf* Vh = Vt + (((size_t)b * NH + head) * HD) * NKv;
  float* sc = smem + (size_t)wave * 16 * NKv;   // wave-private score tile

  // Q A-fragments, d=48 zero-padded to 64
  FragB qa0, qa1;
  {
    const bf* qr = Qh + (size_t)(q0 + lo) * HD;
    qa0.u[0] = *(const u32x4*)(qr + (lhi << 4));
    qa0.u[1] = *(const u32x4*)(qr + (lhi << 4) + 8);
    qa1.u[0] = (lhi == 0) ? *(const u32x4*)(qr + 32)      // d 32..39
                          : (u32x4)0;                     // d 48..55 pad
    qa1.u[1] = (lhi == 0) ? *(const u32x4*)(qr + 40)      // d 40..47
                          : (u32x4)0;                     // d 56..63 pad
  }
  // scores
  for (int kk = 0; kk < NKv; kk += 16) {
    FragB b0, b1;
    const bf* kr = Kh + (size_t)(kk + lo) * HD + (lhi << 4);
    b0.u[0] = *(const u32x4*)(kr);
    b0.u[1] = *(const u32x4*)(kr + 8);
    if (lhi == 0) {                       // k-dim 32..47 (valid half of chunk 1)
      b1.u[0] = *(const u32x4*)(kr + 32);
      b1.u[1] = *(const u32x4*)(kr + 40);
    } else {                              // k-dim 48..63 -> pad
      b1.u[0] = (u32x4)0; b1.u[1] = (u32x4)0;
    }
    v8f c = {};
    c = wmma_bf16(qa0.v, b0.v, c);
    c = wmma_bf16(qa1.v, b1.v, c);
#pragma unroll
    for (int i = 0; i < 8; ++i)
      sc[(size_t)(i + (lhi << 3)) * NKv + kk + lo] = c[i] * scale;
  }
  // softmax over each of the 16 rows (wave-wide)
  for (int m = 0; m < 16; ++m) {
    float* r = sc + (size_t)m * NKv;
    float mx = -3.0e38f;
    for (int j = lane; j < NKv; j += 32) mx = fmaxf(mx, r[j]);
    for (int o = 16; o; o >>= 1) mx = fmaxf(mx, __shfl_xor(mx, o, 32));
    float s = 0.f;
    for (int j = lane; j < NKv; j += 32) { float e = __expf(r[j] - mx); r[j] = e; s += e; }
    for (int o = 16; o; o >>= 1) s += __shfl_xor(s, o, 32);
    float inv = 1.f / s;
    for (int j = lane; j < NKv; j += 32) r[j] *= inv;
  }
  // O = P @ V  (three 16-wide d tiles)
  v8f c0 = {}, c1 = {}, c2 = {};
  for (int k0 = 0; k0 < NKv; k0 += 32) {
    FragB p;
    const float* pr = sc + (size_t)lo * NKv;
#pragma unroll
    for (int e = 0; e < 8; ++e) {
      int k1 = k0 + (lhi << 4) + e;
      int k2 = k0 + (lhi << 4) + 8 + e;
      p.e[e]     = (bf)(k1 < NKv ? pr[k1] : 0.f);
      p.e[e + 8] = (bf)(k2 < NKv ? pr[k2] : 0.f);
    }
    int krow = k0 + (lhi << 4);
    bool ok = (krow + 15) < NKv;          // 784 = 24*32 + 16 tail
#pragma unroll
    for (int dt = 0; dt < 3; ++dt) {
      FragB vb;
      if (ok) {
        const bf* vr = Vh + (size_t)(dt * 16 + lo) * NKv + krow;
        vb.u[0] = *(const u32x4*)(vr);
        vb.u[1] = *(const u32x4*)(vr + 8);
      } else { vb.u[0] = (u32x4)0; vb.u[1] = (u32x4)0; }
      if (dt == 0)      c0 = wmma_bf16(p.v, vb.v, c0);
      else if (dt == 1) c1 = wmma_bf16(p.v, vb.v, c1);
      else              c2 = wmma_bf16(p.v, vb.v, c2);
    }
  }
#pragma unroll
  for (int i = 0; i < 8; ++i) {
    int row = q0 + i + (lhi << 3);
    bf* orow = O + ((size_t)b * NQ + row) * Cc + head * HD;
    orow[lo]      = (bf)c0[i];
    orow[16 + lo] = (bf)c1[i];
    orow[32 + lo] = (bf)c2[i];
  }
}

// ---------------------------------------------------------------------------
extern "C" void kernel_launch(void* const* d_in, const int* in_sizes, int n_in,
                              void* d_out, int out_size, void* d_ws, size_t ws_size,
                              hipStream_t stream) {
  constexpr int Bn = 8, Hh = 56, Ww = 56, Cc = 384;
  constexpr int Nq = Hh * Ww;              // 3136
  constexpr int Nk = (Hh / 2) * (Ww / 2);  // 784
  constexpr int NH = 8, HD = Cc / NH;      // 48
  constexpr int HID = 768;
  constexpr int Mq = Bn * Nq;              // 25088 = 128*196
  constexpr int Mk = Bn * Nk;              // 6272  = 128*49

  const float* x     = (const float*)d_in[0];
  const float* ln1w  = (const float*)d_in[1];
  const float* ln1b  = (const float*)d_in[2];
  const float* qw    = (const float*)d_in[3];
  const float* qb    = (const float*)d_in[4];
  const float* kw    = (const float*)d_in[5];
  const float* kb_   = (const float*)d_in[6];
  const float* vw    = (const float*)d_in[7];
  const float* vb    = (const float*)d_in[8];
  const float* pw    = (const float*)d_in[9];
  const float* pb    = (const float*)d_in[10];
  const float* ln2w  = (const float*)d_in[11];
  const float* ln2b  = (const float*)d_in[12];
  const float* fc1w  = (const float*)d_in[13];
  const float* fc1b  = (const float*)d_in[14];
  const float* fc2w  = (const float*)d_in[15];
  const float* fc2b  = (const float*)d_in[16];
  float* out = (float*)d_out;

  // workspace carve
  char* p = (char*)d_ws;
  auto take = [&](size_t bytes) { char* r = p; p += (bytes + 255) & ~size_t(255); return r; };
  bf* qw_t  = (bf*)take((size_t)Cc * Cc * 2);
  bf* kw_t  = (bf*)take((size_t)Cc * Cc * 2);
  bf* vw_t  = (bf*)take((size_t)Cc * Cc * 2);
  bf* pw_t  = (bf*)take((size_t)Cc * Cc * 2);
  bf* f1_t  = (bf*)take((size_t)Cc * HID * 2);
  bf* f2_t  = (bf*)take((size_t)HID * Cc * 2);
  bf* qtok  = (bf*)take((size_t)Mq * Cc * 2);
  bf* kvtok = (bf*)take((size_t)Mk * Cc * 2);
  bf* Qb    = (bf*)take((size_t)Mq * Cc * 2);
  bf* Kb    = (bf*)take((size_t)Mk * Cc * 2);
  bf* Vtb   = (bf*)take((size_t)Mk * Cc * 2);
  bf* Ob    = (bf*)take((size_t)Mq * Cc * 2);
  float* x1 = (float*)take((size_t)Mq * Cc * 4);
  bf* yb    = (bf*)take((size_t)Mq * Cc * 2);
  bf* hact  = (bf*)take((size_t)Mq * HID * 2);
  (void)ws_size; (void)n_in; (void)in_sizes; (void)out_size;

  // 1. weights -> bf16, transposed to [N][K]
  k_cvt_transpose<<<(Cc * Cc + 255) / 256, 256, 0, stream>>>(qw, qw_t, Cc, Cc);
  k_cvt_transpose<<<(Cc * Cc + 255) / 256, 256, 0, stream>>>(kw, kw_t, Cc, Cc);
  k_cvt_transpose<<<(Cc * Cc + 255) / 256, 256, 0, stream>>>(vw, vw_t, Cc, Cc);
  k_cvt_transpose<<<(Cc * Cc + 255) / 256, 256, 0, stream>>>(pw, pw_t, Cc, Cc);
  k_cvt_transpose<<<(Cc * HID + 255) / 256, 256, 0, stream>>>(fc1w, f1_t, Cc, HID);
  k_cvt_transpose<<<(HID * Cc + 255) / 256, 256, 0, stream>>>(fc2w, f2_t, HID, Cc);

  // 2. LN1(x) -> bf16 q tokens ; pooled raw x -> bf16 kv tokens
  k_layernorm_bf16<<<Mq, 128, 0, stream>>>(x, ln1w, ln1b, qtok, Cc);
  k_pool_bf16<<<(Mk * Cc + 255) / 256, 256, 0, stream>>>(x, kvtok, Bn, Hh, Ww, Cc);

  auto gemm_blocks = [](int M, int N) { return (M / 128) * (N / 64); };

  // 3. projections (WMMA, async-LDS B panels)
  k_gemm_wmma<<<gemm_blocks(Mq, Cc), 256, 0, stream>>>(
      qtok, qw_t, qb, nullptr, nullptr, Qb, Mq, Cc, Cc, /*layout*/1, Nq, HD, 0);
  k_gemm_wmma<<<gemm_blocks(Mk, Cc), 256, 0, stream>>>(
      kvtok, kw_t, kb_, nullptr, nullptr, Kb, Mk, Cc, Cc, /*layout*/1, Nk, HD, 0);
  k_gemm_wmma<<<gemm_blocks(Mk, Cc), 256, 0, stream>>>(
      kvtok, vw_t, vb, nullptr, nullptr, Vtb, Mk, Cc, Cc, /*layout*/2, Nk, HD, 0);

  // 4. attention (WMMA scores + WMMA PV)
  {
    dim3 grid(Nq / 32, NH, Bn);
    size_t shmem = (size_t)2 * 16 * Nk * sizeof(float);
    float scale = 0.14433756729740643f; // 48^-0.5
    k_attn_wmma<<<grid, 64, shmem, stream>>>(Qb, Kb, Vtb, Ob, Nq, Nk, HD, NH, Cc, scale);
  }

  // 5. out proj + residual -> x1 (f32)
  k_gemm_wmma<<<gemm_blocks(Mq, Cc), 256, 0, stream>>>(
      Ob, pw_t, pb, x, x1, nullptr, Mq, Cc, Cc, 0, Nq, HD, 0);

  // 6. MLP
  k_layernorm_bf16<<<Mq, 128, 0, stream>>>(x1, ln2w, ln2b, yb, Cc);
  k_gemm_wmma<<<gemm_blocks(Mq, HID), 256, 0, stream>>>(
      yb, f1_t, fc1b, nullptr, nullptr, hact, Mq, HID, Cc, 0, Nq, HD, /*gelu*/1);
  k_gemm_wmma<<<gemm_blocks(Mq, Cc), 256, 0, stream>>>(
      hact, f2_t, fc2b, x1, out, nullptr, Mq, Cc, HID, 0, Nq, HD, 0);
}
